// ClipRISV33_41540923687031
// MI455X (gfx1250) — matrix operand order repaired and checked by reading, not compiled, not run on hardware
//
#include <hip/hip_runtime.h>
#include <math.h>

typedef __attribute__((ext_vector_type(16))) _Float16 v16h;
typedef __attribute__((ext_vector_type(8)))  float    v8f;

#define CDIV(a, b) (((a) + (b) - 1) / (b))

// ---------------------------------------------------------------------------
// WMMA helpers (CDNA5 wave32, 16x16x32 f16 -> f32 accumulate)
// ---------------------------------------------------------------------------
__device__ __forceinline__ v8f wmma16(v16h a, v16h b, v8f c) {
  return __builtin_amdgcn_wmma_f32_16x16x32_f16(
      false, a, false, b, (short)0, c, false, false);
}

// A-fragment (M=16, K=32, f16). Lanes 0-15: row M=lane, K 0..15;
// lanes 16-31: row M=lane-16, K 16..31. f(m,k)->_Float16.
template <typename F>
__device__ __forceinline__ v16h frag_a(int lane, int kbase, F f) {
  union { v16h v; _Float16 h[16]; } u;
  const int m  = lane & 15;
  const int k0 = kbase + ((lane >> 4) << 4);
#pragma unroll
  for (int e = 0; e < 16; ++e)
    u.h[e] = f(m, k0 + e);
  return u.v;
}

// B-fragment (K=32, N=16, f16). Lane = col N (mod 16); lanes 0-15 hold
// K=0..15, lanes 16-31 hold K=16..31. f(k,n)->_Float16. K consecutive in e.
template <typename F>
__device__ __forceinline__ v16h frag_b(int lane, int kbase, F f) {
  union { v16h v; _Float16 h[16]; } u;
  const int n  = lane & 15;
  const int k0 = kbase + ((lane >> 4) << 4);
#pragma unroll
  for (int e = 0; e < 16; ++e) u.h[e] = f(k0 + e, n);
  return u.v;
}

// B-fragment from LDS, channel-innermost: 16 consecutive halves per lane.
__device__ __forceinline__ v16h frag_b_lds(const _Float16* src) {
  union { v16h v; _Float16 h[16]; } u;
#pragma unroll
  for (int e = 0; e < 16; ++e) u.h[e] = src[e];
  return u.v;
}

// C/D 16x16 f32: lane = N column; VGPR r = row M (lanes 16-31 -> M=8+r).
template <typename F>
__device__ __forceinline__ void frag_store(int lane, v8f c, F f) {
  const int n  = lane & 15;
  const int m0 = (lane >> 4) << 3;
#pragma unroll
  for (int r = 0; r < 8; ++r) f(m0 + r, n, c[r]);
}

// ---------------------------------------------------------------------------
// Prep / elementwise kernels
// ---------------------------------------------------------------------------

// Pack conv weights [OC,IC,3,3] f32 -> f16, K-order k=(cc*9+tap)*32+ci,
// channel c = cc*32+ci zero-padded to ICpad.
__global__ void k_pack_w(_Float16* dst, const float* src, int OC, int IC, int ICpad) {
  int i = blockIdx.x * 256 + threadIdx.x;
  int K = ICpad * 9;
  if (i >= OC * K) return;
  int oc = i / K, k = i - oc * K;
  int cc = k / 288, r = k - cc * 288, tap = r >> 5, ci = r & 31;
  int c = cc * 32 + ci;
  dst[i] = (c < IC) ? (_Float16)src[(oc * IC + c) * 9 + tap] : (_Float16)0.f;
}

// transpose [K,N] f32 -> [N,K] f16 (K-contiguous rows for B fragments)
__global__ void k_f2h_t(_Float16* dst, const float* src, int K, int N) {
  int i = blockIdx.x * 256 + threadIdx.x;
  if (i >= K * N) return;
  int n = i / K, k = i - n * K;
  dst[i] = (_Float16)src[k * N + n];
}

__global__ void k_hzero(_Float16* dst, int n) {
  int i = blockIdx.x * 256 + threadIdx.x;
  if (i < n) dst[i] = (_Float16)0.f;
}

// tok512 [616,512] f32 -> [624,512] f16 zero-padded rows
__global__ void k_tokpad(_Float16* dst, const float* src) {
  int i = blockIdx.x * 256 + threadIdx.x;
  if (i >= 624 * 512) return;
  int row = i >> 9;
  dst[i] = (row < 616) ? (_Float16)src[i] : (_Float16)0.f;
}

// conv1: 3->64, 3x3, stride2, pad1, relu. out f16 [8,64,256,256]
__global__ void k_conv1(_Float16* out, const float* img, const float* w, const float* bias) {
  int i = blockIdx.x * 256 + threadIdx.x;
  if (i >= 8 * 64 * 256 * 256) return;
  int x = i & 255, y = (i >> 8) & 255, oc = (i >> 16) & 63, b = i >> 22;
  float acc = bias[oc];
  for (int ic = 0; ic < 3; ++ic)
    for (int ky = 0; ky < 3; ++ky) {
      int iy = 2 * y - 1 + ky;
      if ((unsigned)iy >= 512u) continue;
      for (int kx = 0; kx < 3; ++kx) {
        int ix = 2 * x - 1 + kx;
        if ((unsigned)ix >= 512u) continue;
        acc += img[((b * 3 + ic) * 512 + iy) * 512 + ix] * w[((oc * 3 + ic) * 3 + ky) * 3 + kx];
      }
    }
  out[i] = (_Float16)fmaxf(acc, 0.f);
}

__global__ void k_hid(float* hid, const float* w, const float* b) {
  int i = blockIdx.x * 256 + threadIdx.x;
  if (i >= 64 * 16384) return;
  int p = i & 16383, j = i >> 14;
  int x = p & 127, y = p >> 7;
  float xv = -1.f + 2.f * (float)x / 127.f;
  float yv = -1.f + 2.f * (float)y / 127.f;
  hid[i] = fmaxf(w[j * 2] * xv + w[j * 2 + 1] * yv + b[j], 0.f);
}

__global__ void k_pos(float* pos, const float* hid, const float* w, const float* b) {
  int i = blockIdx.x * 256 + threadIdx.x;
  if (i >= 256 * 16384) return;
  int p = i & 16383, c = i >> 14;
  float s = b[c];
  for (int j = 0; j < 64; ++j) s += w[c * 64 + j] * hid[j * 16384 + p];
  pos[i] = s;
}

// ---------------------------------------------------------------------------
// LDS-staged implicit-GEMM conv kernels
// ---------------------------------------------------------------------------

// conv2: 64->256, 3x3, stride2, pad1, on 256x256 input -> 128x128.
// Block: (b, 64-px group, mhalf). 8 waves = 8 consecutive out-channel tiles.
// LDS tile per 32-ch chunk: [3 rows][130 px][32 ch] halves.
__global__ void k_conv2(float* img, const _Float16* xin, const _Float16* wA,
                        const float* bias, const float* pos) {
  __shared__ _Float16 tile[3 * 130 * 32];
  const int tid = threadIdx.x, lane = tid & 31, wid = tid >> 5;
  const int blk = blockIdx.x;             // 8 * 256 * 2
  const int mhalf = blk & 1, pg = (blk >> 1) & 255, b = blk >> 9;
  const int y0 = pg >> 1, x0 = (pg & 1) * 64;
  const int mt = mhalf * 8 + wid;
  v8f acc[4] = {};
  for (int cc = 0; cc < 2; ++cc) {
    __syncthreads();
    for (int idx = tid; idx < 3 * 130 * 32; idx += 256) {
      int lx = idx % 130;
      int rest = idx / 130;
      int ci = rest & 31, row = rest >> 5;
      int iy = 2 * y0 - 1 + row, ix = 2 * x0 - 1 + lx;
      _Float16 v = (_Float16)0.f;
      if ((unsigned)iy < 256u && (unsigned)ix < 256u) {
        v = xin[((b * 64 + cc * 32 + ci) << 16) + (iy << 8) + ix];
        if (cc == 0)
          __builtin_prefetch(&xin[((b * 64 + 32 + ci) << 16) + (iy << 8) + ix], 0, 1);
      }
      tile[(row * 130 + lx) * 32 + ci] = v;
    }
    __syncthreads();
#pragma unroll
    for (int tap = 0; tap < 9; ++tap) {
      const int ky = tap / 3, kx = tap - ky * 3;
      const int kbase = (cc * 9 + tap) * 32;
      v16h a = frag_a(lane, kbase, [&](int m, int k) -> _Float16 {
        return wA[(mt * 16 + m) * 576 + k];
      });
      const int ci0 = (lane >> 4) << 4;
      const int n = lane & 15;
#pragma unroll
      for (int t = 0; t < 4; ++t) {
        int lx = 2 * (t * 16 + n) + kx;
        v16h bb = frag_b_lds(&tile[(ky * 130 + lx) * 32 + ci0]);
        acc[t] = wmma16(a, bb, acc[t]);
      }
    }
  }
  const int pbase = pg * 64;
#pragma unroll
  for (int t = 0; t < 4; ++t)
    frag_store(lane, acc[t], [&](int m, int n, float v) {
      int oc = mt * 16 + m, p = pbase + t * 16 + n;
      img[((b * 256 + oc) << 14) + p] = fmaxf(v + bias[oc], 0.f) + pos[(oc << 14) + p];
    });
}

// d1: 544->256, 3x3, stride1, pad1 on 128x128. 17 channel chunks.
__global__ void k_d1(float* y, const _Float16* fin, const _Float16* wA, const float* bias) {
  __shared__ _Float16 tile[3 * 66 * 32];
  const int tid = threadIdx.x, lane = tid & 31, wid = tid >> 5;
  const int blk = blockIdx.x;             // 8 * 256 * 2
  const int mhalf = blk & 1, pg = (blk >> 1) & 255, b = blk >> 9;
  const int y0 = pg >> 1, x0 = (pg & 1) * 64;
  const int mt = mhalf * 8 + wid;
  v8f acc[4] = {};
  for (int cc = 0; cc < 17; ++cc) {
    __syncthreads();
    for (int idx = tid; idx < 3 * 66 * 32; idx += 256) {
      int lx = idx % 66;
      int rest = idx / 66;
      int ci = rest & 31, row = rest >> 5;
      int iy = y0 - 1 + row, ix = x0 - 1 + lx;
      _Float16 v = (_Float16)0.f;
      if ((unsigned)iy < 128u && (unsigned)ix < 128u) {
        v = fin[((b * 544 + cc * 32 + ci) << 14) + (iy << 7) + ix];
        if (cc < 16)
          __builtin_prefetch(&fin[((b * 544 + cc * 32 + 32 + ci) << 14) + (iy << 7) + ix], 0, 1);
      }
      tile[(row * 66 + lx) * 32 + ci] = v;
    }
    __syncthreads();
#pragma unroll
    for (int tap = 0; tap < 9; ++tap) {
      const int ky = tap / 3, kx = tap - ky * 3;
      const int kbase = (cc * 9 + tap) * 32;
      v16h a = frag_a(lane, kbase, [&](int m, int k) -> _Float16 {
        return wA[(mt * 16 + m) * 4896 + k];
      });
      const int ci0 = (lane >> 4) << 4;
      const int n = lane & 15;
#pragma unroll
      for (int t = 0; t < 4; ++t) {
        int lx = t * 16 + n + kx;
        v16h bb = frag_b_lds(&tile[(ky * 66 + lx) * 32 + ci0]);
        acc[t] = wmma16(a, bb, acc[t]);
      }
    }
  }
  const int pbase = pg * 64;
#pragma unroll
  for (int t = 0; t < 4; ++t)
    frag_store(lane, acc[t], [&](int m, int n, float v) {
      int oc = mt * 16 + m, p = pbase + t * 16 + n;
      y[((b * 256 + oc) << 14) + p] = v + bias[oc];
    });
}

// d2: 256->64, 3x3, stride1, pad1, relu. 8 chunks; wave = (mt, pixel half).
__global__ void k_d2(float* y, const _Float16* xin, const _Float16* wA, const float* bias) {
  __shared__ _Float16 tile[3 * 66 * 32];
  const int tid = threadIdx.x, lane = tid & 31, wid = tid >> 5;
  const int blk = blockIdx.x;             // 8 * 256
  const int pg = blk & 255, b = blk >> 8;
  const int y0 = pg >> 1, x0 = (pg & 1) * 64;
  const int mt = wid & 3, tset = wid >> 2;
  v8f acc[2] = {};
  for (int cc = 0; cc < 8; ++cc) {
    __syncthreads();
    for (int idx = tid; idx < 3 * 66 * 32; idx += 256) {
      int lx = idx % 66;
      int rest = idx / 66;
      int ci = rest & 31, row = rest >> 5;
      int iy = y0 - 1 + row, ix = x0 - 1 + lx;
      _Float16 v = (_Float16)0.f;
      if ((unsigned)iy < 128u && (unsigned)ix < 128u) {
        v = xin[((b * 256 + cc * 32 + ci) << 14) + (iy << 7) + ix];
        if (cc < 7)
          __builtin_prefetch(&xin[((b * 256 + cc * 32 + 32 + ci) << 14) + (iy << 7) + ix], 0, 1);
      }
      tile[(row * 66 + lx) * 32 + ci] = v;
    }
    __syncthreads();
#pragma unroll
    for (int tap = 0; tap < 9; ++tap) {
      const int ky = tap / 3, kx = tap - ky * 3;
      const int kbase = (cc * 9 + tap) * 32;
      v16h a = frag_a(lane, kbase, [&](int m, int k) -> _Float16 {
        return wA[(mt * 16 + m) * 2304 + k];
      });
      const int ci0 = (lane >> 4) << 4;
      const int n = lane & 15;
#pragma unroll
      for (int tt = 0; tt < 2; ++tt) {
        int lx = (tset * 2 + tt) * 16 + n + kx;
        v16h bb = frag_b_lds(&tile[(ky * 66 + lx) * 32 + ci0]);
        acc[tt] = wmma16(a, bb, acc[tt]);
      }
    }
  }
  const int pbase = pg * 64;
#pragma unroll
  for (int tt = 0; tt < 2; ++tt)
    frag_store(lane, acc[tt], [&](int m, int n, float v) {
      int oc = mt * 16 + m, p = pbase + (tset * 2 + tt) * 16 + n;
      y[((b * 64 + oc) << 14) + p] = fmaxf(v + bias[oc], 0.f);
    });
}

// ---------------------------------------------------------------------------
// Dense GEMMs (attention + token path)
// ---------------------------------------------------------------------------

// per-pixel channel L2 norm -> dense_h f16 [B,HW,256]
__global__ void k_dense(_Float16* dh, const float* img) {
  int lane = threadIdx.x & 31;
  int pg = (blockIdx.x << 3) + (threadIdx.x >> 5);
  if (pg >= 8 * 16384) return;
  int b = pg >> 14, p = pg & 16383;
  float v[8], ss = 0.f;
#pragma unroll
  for (int j = 0; j < 8; ++j) {
    int c = lane + (j << 5);
    v[j] = img[((b * 256 + c) << 14) + p];
    ss += v[j] * v[j];
  }
#pragma unroll
  for (int o = 16; o > 0; o >>= 1) ss += __shfl_xor(ss, o, 32);
  float inv = 1.f / fmaxf(sqrtf(ss), 1e-12f);
#pragma unroll
  for (int j = 0; j < 8; ++j) {
    int c = lane + (j << 5);
    dh[((b << 14) + p) * 256 + c] = (_Float16)(v[j] * inv);
  }
}

// proj GEMM: [624,512] x W^T[256,512] + b
__global__ void k_proj(float* pf, _Float16* ph, const _Float16* A, const _Float16* BwT,
                       const float* bias) {
  int lane = threadIdx.x & 31;
  int w = (blockIdx.x << 3) + (threadIdx.x >> 5);
  if (w >= 39 * 16) return;
  int mt = w >> 4, nt = w & 15;
  v8f acc = {};
  for (int kk = 0; kk < 16; ++kk) {
    v16h a = frag_a(lane, kk * 32, [&](int m, int k) -> _Float16 {
      return A[(mt * 16 + m) * 512 + k];
    });
    v16h bb = frag_b(lane, kk * 32, [&](int k, int n) -> _Float16 {
      return BwT[(nt * 16 + n) * 512 + k];
    });
    acc = wmma16(a, bb, acc);
  }
  frag_store(lane, acc, [&](int m, int n, float v) {
    int row = mt * 16 + m, col = nt * 16 + n;
    float r = v + bias[col];
    bool ok = row < 616;
    pf[row * 256 + col] = ok ? r : 0.f;
    ph[row * 256 + col] = ok ? (_Float16)r : (_Float16)0.f;
  });
}

// gate GEMM: sigmoid(proj@gate_w + b) * proj
__global__ void k_gate(float* traw, const _Float16* A, const _Float16* BwT,
                       const float* bias, const float* pf) {
  int lane = threadIdx.x & 31;
  int w = (blockIdx.x << 3) + (threadIdx.x >> 5);
  if (w >= 39 * 16) return;
  int mt = w >> 4, nt = w & 15;
  v8f acc = {};
  for (int kk = 0; kk < 8; ++kk) {
    v16h a = frag_a(lane, kk * 32, [&](int m, int k) -> _Float16 {
      return A[(mt * 16 + m) * 256 + k];
    });
    v16h bb = frag_b(lane, kk * 32, [&](int k, int n) -> _Float16 {
      return BwT[(nt * 16 + n) * 256 + k];
    });
    acc = wmma16(a, bb, acc);
  }
  frag_store(lane, acc, [&](int m, int n, float v) {
    int row = mt * 16 + m, col = nt * 16 + n;
    float g = 1.f / (1.f + __expf(-(v + bias[col])));
    traw[row * 256 + col] = (row < 616) ? g * pf[row * 256 + col] : 0.f;
  });
}

// token L2 norm -> tn f32, th f16 [B,96,256], tokT f16 [B,256,96]
__global__ void k_toknorm(float* tn, _Float16* th, _Float16* tokT, const float* traw) {
  int lane = threadIdx.x & 31;
  int row = (blockIdx.x << 3) + (threadIdx.x >> 5);
  if (row >= 616) return;
  float v[8], ss = 0.f;
#pragma unroll
  for (int j = 0; j < 8; ++j) {
    int c = lane + (j << 5);
    v[j] = traw[row * 256 + c];
    ss += v[j] * v[j];
  }
#pragma unroll
  for (int o = 16; o > 0; o >>= 1) ss += __shfl_xor(ss, o, 32);
  float inv = 1.f / fmaxf(sqrtf(ss), 1e-12f);
  int b = row / 77, l = row - b * 77;
#pragma unroll
  for (int j = 0; j < 8; ++j) {
    int c = lane + (j << 5);
    float r = v[j] * inv;
    tn[row * 256 + c] = r;
    th[((b * 96 + l) << 8) + c] = (_Float16)r;
    tokT[((b << 8) + c) * 96 + l] = (_Float16)r;
  }
}

__global__ void k_gvec(float* gvec, const float* tn, const int* toks) {
  int b = blockIdx.x, c = threadIdx.x;
  int cnt = 0;
  float s = 0.f;
  for (int l = 0; l < 77; ++l) {
    if (toks[b * 77 + l] != 0) { cnt++; s += tn[(b * 77 + l) * 256 + c]; }
  }
  float v = s / fmaxf((float)cnt, 1.f);
  __shared__ float red[256];
  red[c] = v * v;
  __syncthreads();
  for (int o = 128; o > 0; o >>= 1) {
    if (c < o) red[c] += red[c + o];
    __syncthreads();
  }
  gvec[b * 256 + c] = v / fmaxf(sqrtf(red[0]), 1e-12f);
}

// scores GEMM: dense [HW,256] x tok^T -> [HW,96], scaled + masked
__global__ void k_scores(float* sc, const _Float16* dh, const _Float16* th, const int* toks) {
  int lane = threadIdx.x & 31;
  int w = (blockIdx.x << 3) + (threadIdx.x >> 5);
  if (w >= 8 * 1024) return;
  int b = w >> 10, mt = w & 1023;
  v8f acc[6] = {};
  for (int kk = 0; kk < 8; ++kk) {
    v16h a = frag_a(lane, kk * 32, [&](int m, int k) -> _Float16 {
      return dh[((b << 14) + mt * 16 + m) * 256 + k];
    });
#pragma unroll
    for (int nt = 0; nt < 6; ++nt) {
      v16h bb = frag_b(lane, kk * 32, [&](int k, int n) -> _Float16 {
        return th[((b * 96 + nt * 16 + n) << 8) + k];
      });
      acc[nt] = wmma16(a, bb, acc[nt]);
    }
  }
#pragma unroll
  for (int nt = 0; nt < 6; ++nt)
    frag_store(lane, acc[nt], [&](int m, int n, float v) {
      int p = mt * 16 + m, l = nt * 16 + n;
      float r;
      if (l >= 77) r = -1e30f;
      else r = (toks[b * 77 + l] != 0) ? v * 0.0625f : -1e4f;
      sc[((b << 14) + p) * 96 + l] = r;
    });
}

__global__ void k_softmax(_Float16* at, const float* sc) {
  int i = blockIdx.x * 256 + threadIdx.x;
  if (i >= 8 * 16384) return;
  const float* row = sc + (long)i * 96;
  float mx = -1e30f;
  for (int l = 0; l < 96; ++l) mx = fmaxf(mx, row[l]);
  float s = 0.f;
  for (int l = 0; l < 96; ++l) s += __expf(row[l] - mx);
  float inv = 1.f / s;
  for (int l = 0; l < 96; ++l) at[(long)i * 96 + l] = (_Float16)(__expf(row[l] - mx) * inv);
}

// ctx GEMM: attn [HW,96] x tok [96,256] -> fused channels 256..511
__global__ void k_ctx(_Float16* fused, const _Float16* at, const _Float16* tokT) {
  int lane = threadIdx.x & 31;
  int w = (blockIdx.x << 3) + (threadIdx.x >> 5);
  if (w >= 8 * 1024 * 4) return;
  int b = w >> 12, mt = (w >> 2) & 1023, ng = w & 3;
  v8f acc[4] = {};
  for (int kk = 0; kk < 3; ++kk) {
    v16h a = frag_a(lane, kk * 32, [&](int m, int k) -> _Float16 {
      return at[((b << 14) + mt * 16 + m) * 96 + k];
    });
#pragma unroll
    for (int t = 0; t < 4; ++t) {
      int nb = (ng * 4 + t) * 16;
      v16h bb = frag_b(lane, kk * 32, [&](int k, int n) -> _Float16 {
        return tokT[((b << 8) + nb + n) * 96 + k];
      });
      acc[t] = wmma16(a, bb, acc[t]);
    }
  }
#pragma unroll
  for (int t = 0; t < 4; ++t)
    frag_store(lane, acc[t], [&](int m, int n, float v) {
      int p = mt * 16 + m, c = (ng * 4 + t) * 16 + n;
      fused[((b * 544 + 256 + c) << 14) + p] = (_Float16)v;
    });
}

__global__ void k_sim(float* sim, const _Float16* dh, const float* gvec) {
  int lane = threadIdx.x & 31;
  int pg = (blockIdx.x << 3) + (threadIdx.x >> 5);
  if (pg >= 8 * 16384) return;
  int b = pg >> 14, p = pg & 16383;
  float s = 0.f;
#pragma unroll
  for (int j = 0; j < 8; ++j) {
    int c = lane + (j << 5);
    s += (float)dh[((b << 14) + p) * 256 + c] * gvec[b * 256 + c];
  }
#pragma unroll
  for (int o = 16; o > 0; o >>= 1) s += __shfl_xor(s, o, 32);
  if (lane == 0) sim[(b << 14) + p] = s;
}

__global__ void k_fused(_Float16* fused, const float* img, const float* sim) {
  int i = blockIdx.x * 256 + threadIdx.x;
  if (i >= 8 * 544 * 16384) return;
  int p = i & 16383, c = (i >> 14) % 544, b = i / (544 * 16384);
  if (c < 256) fused[i] = (_Float16)img[((b * 256 + c) << 14) + p];
  else if (c == 512) fused[i] = (_Float16)sim[(b << 14) + p];
  else if (c > 512) fused[i] = (_Float16)0.f;
}

__global__ void k_gn(_Float16* out, const float* y, const float* g, const float* bta) {
  int bg = blockIdx.x;
  int b = bg >> 5, grp = bg & 31;
  int tid = threadIdx.x;
  const float* base = y + ((b * 256 + grp * 8) << 14);
  float s = 0.f, ss = 0.f;
  for (int i = tid; i < 8 * 16384; i += 256) {
    float v = base[i];
    s += v;
    ss += v * v;
  }
  __shared__ float r1[256], r2[256];
  r1[tid] = s; r2[tid] = ss;
  __syncthreads();
  for (int o = 128; o > 0; o >>= 1) {
    if (tid < o) { r1[tid] += r1[tid + o]; r2[tid] += r2[tid + o]; }
    __syncthreads();
  }
  float mean = r1[0] * (1.f / 131072.f);
  float var = r2[0] * (1.f / 131072.f) - mean * mean;
  float inv = rsqrtf(var + 1e-5f);
  for (int i = tid; i < 8 * 16384; i += 256) {
    int c = grp * 8 + (i >> 14);
    float v = (base[i] - mean) * inv * g[c] + bta[c];
    out[((b * 256 + grp * 8) << 14) + i] = (_Float16)fmaxf(v, 0.f);
  }
}

__global__ void k_d3(float* y3, const float* y2, const float* w, const float* b3) {
  int i = blockIdx.x * 256 + threadIdx.x;
  if (i >= 8 * 16384) return;
  int b = i >> 14, p = i & 16383;
  float s = b3[0];
  for (int c = 0; c < 64; ++c) s += y2[((b * 64 + c) << 14) + p] * w[c];
  y3[i] = s;
}

__global__ void k_up(float* out, const float* y3) {
  int i = blockIdx.x * 256 + threadIdx.x;
  if (i >= 8 * 512 * 512) return;
  int X = i & 511, Y = (i >> 9) & 511, b = i >> 18;
  float sx = (X + 0.5f) * 0.25f - 0.5f;
  float sy = (Y + 0.5f) * 0.25f - 0.5f;
  int x0 = (int)floorf(sx), y0 = (int)floorf(sy);
  float fx = sx - (float)x0, fy = sy - (float)y0;
  int x0c = min(max(x0, 0), 127), x1c = min(max(x0 + 1, 0), 127);
  int y0c = min(max(y0, 0), 127), y1c = min(max(y0 + 1, 0), 127);
  const float* src = y3 + (b << 14);
  float v00 = src[(y0c << 7) + x0c], v01 = src[(y0c << 7) + x1c];
  float v10 = src[(y1c << 7) + x0c], v11 = src[(y1c << 7) + x1c];
  out[i] = (1.f - fy) * ((1.f - fx) * v00 + fx * v01) + fy * ((1.f - fx) * v10 + fx * v11);
}

// ---------------------------------------------------------------------------
// Host launch
// ---------------------------------------------------------------------------
extern "C" void kernel_launch(void* const* d_in, const int* in_sizes, int n_in,
                              void* d_out, int out_size, void* d_ws, size_t ws_size,
                              hipStream_t stream) {
  (void)in_sizes; (void)n_in; (void)out_size; (void)ws_size;
  const float* images = (const float*)d_in[0];
  const int*   toks   = (const int*)d_in[1];
  const float* tok512 = (const float*)d_in[2];
  const float* c1w    = (const float*)d_in[3];
  const float* c1b    = (const float*)d_in[4];
  const float* c2w    = (const float*)d_in[5];
  const float* c2b    = (const float*)d_in[6];
  const float* sp1w   = (const float*)d_in[7];
  const float* sp1b   = (const float*)d_in[8];
  const float* sp2w   = (const float*)d_in[9];
  const float* sp2b   = (const float*)d_in[10];
  const float* projw  = (const float*)d_in[11];
  const float* projb  = (const float*)d_in[12];
  const float* gatew  = (const float*)d_in[13];
  const float* gateb  = (const float*)d_in[14];
  const float* d1w    = (const float*)d_in[15];
  const float* d1b    = (const float*)d_in[16];
  const float* gng    = (const float*)d_in[17];
  const float* gnb    = (const float*)d_in[18];
  const float* d2w    = (const float*)d_in[19];
  const float* d2b    = (const float*)d_in[20];
  const float* d3w    = (const float*)d_in[21];
  const float* d3b    = (const float*)d_in[22];
  float* out = (float*)d_out;

  char* ws = (char*)d_ws;
  size_t off = 0;
  auto alloc = [&](size_t bytes) -> void* {
    void* p = ws + off;
    off += (bytes + 255) & ~(size_t)255;
    return p;
  };

  _Float16* conv1h = (_Float16*)alloc(8UL * 64 * 256 * 256 * 2);
  float*    img    = (float*)   alloc(8UL * 256 * 16384 * 4);
  _Float16* dense  = (_Float16*)alloc(8UL * 16384 * 256 * 2);
  float*    pos    = (float*)   alloc(256UL * 16384 * 4);
  float*    hid    = (float*)   alloc(64UL * 16384 * 4);
  _Float16* wc2    = (_Float16*)alloc(256UL * 576 * 2);
  _Float16* wd1    = (_Float16*)alloc(256UL * 4896 * 2);
  _Float16* wd2    = (_Float16*)alloc(64UL * 2304 * 2);
  _Float16* pwT    = (_Float16*)alloc(256UL * 512 * 2);
  _Float16* gwT    = (_Float16*)alloc(256UL * 256 * 2);
  _Float16* t512   = (_Float16*)alloc(624UL * 512 * 2);
  float*    projf  = (float*)   alloc(624UL * 256 * 4);
  _Float16* projh  = (_Float16*)alloc(624UL * 256 * 2);
  float*    traw   = (float*)   alloc(624UL * 256 * 4);
  float*    tn     = (float*)   alloc(624UL * 256 * 4);
  _Float16* tokh   = (_Float16*)alloc(8UL * 96 * 256 * 2);
  _Float16* tokT   = (_Float16*)alloc(8UL * 256 * 96 * 2);
  float*    gvec   = (float*)   alloc(8UL * 256 * 4);
  float*    scores = (float*)   alloc(8UL * 16384 * 96 * 4);
  _Float16* attnh  = (_Float16*)alloc(8UL * 16384 * 96 * 2);
  float*    sim    = (float*)   alloc(8UL * 16384 * 4);
  _Float16* fused  = (_Float16*)alloc(8UL * 544 * 16384 * 2);
  float*    y3     = (float*)   alloc(8UL * 16384 * 4);
  // aliases over dead buffers
  float*    y1  = img;            // d1 raw out  (img dead after k_fused)
  _Float16* y1h = dense;          // GN out      (dense dead after scores/sim)
  float*    y2  = (float*)conv1h; // d2 out      (conv1h dead after conv2)

  // --- weight prep ---
  k_pack_w<<<CDIV(256 * 576, 256), 256, 0, stream>>>(wc2, c2w, 256, 64, 64);
  k_pack_w<<<CDIV(256 * 4896, 256), 256, 0, stream>>>(wd1, d1w, 256, 513, 544);
  k_pack_w<<<CDIV(64 * 2304, 256), 256, 0, stream>>>(wd2, d2w, 64, 256, 256);
  k_f2h_t<<<CDIV(512 * 256, 256), 256, 0, stream>>>(pwT, projw, 512, 256);
  k_f2h_t<<<CDIV(256 * 256, 256), 256, 0, stream>>>(gwT, gatew, 256, 256);
  k_tokpad<<<CDIV(624 * 512, 256), 256, 0, stream>>>(t512, tok512);
  k_hzero<<<CDIV(8 * 96 * 256, 256), 256, 0, stream>>>(tokh, 8 * 96 * 256);
  k_hzero<<<CDIV(8 * 256 * 96, 256), 256, 0, stream>>>(tokT, 8 * 256 * 96);

  // --- image stem + positional residual ---
  k_conv1<<<CDIV(8 * 64 * 256 * 256, 256), 256, 0, stream>>>(conv1h, images, c1w, c1b);
  k_hid<<<CDIV(64 * 16384, 256), 256, 0, stream>>>(hid, sp1w, sp1b);
  k_pos<<<CDIV(256 * 16384, 256), 256, 0, stream>>>(pos, hid, sp2w, sp2b);
  k_conv2<<<8 * 256 * 2, 256, 0, stream>>>(img, conv1h, wc2, c2b, pos);
  k_dense<<<CDIV(8 * 16384, 8), 256, 0, stream>>>(dense, img);

  // --- token path ---
  k_proj<<<CDIV(39 * 16, 8), 256, 0, stream>>>(projf, projh, t512, pwT, projb);
  k_gate<<<CDIV(39 * 16, 8), 256, 0, stream>>>(traw, projh, gwT, gateb, projf);
  k_toknorm<<<CDIV(616, 8), 256, 0, stream>>>(tn, tokh, tokT, traw);
  k_gvec<<<8, 256, 0, stream>>>(gvec, tn, toks);

  // --- cross attention ---
  k_scores<<<CDIV(8 * 1024, 8), 256, 0, stream>>>(scores, dense, tokh, toks);
  k_softmax<<<CDIV(8 * 16384, 256), 256, 0, stream>>>(attnh, scores);
  k_sim<<<CDIV(8 * 16384, 8), 256, 0, stream>>>(sim, dense, gvec);
  k_fused<<<CDIV(8 * 544 * 16384, 256), 256, 0, stream>>>(fused, img, sim);
  k_ctx<<<CDIV(8 * 1024 * 4, 8), 256, 0, stream>>>(fused, attnh, tokT);

  // --- decoder ---
  k_d1<<<8 * 256 * 2, 256, 0, stream>>>(y1, fused, wd1, d1b);
  k_gn<<<8 * 32, 256, 0, stream>>>(y1h, y1, gng, gnb);
  k_d2<<<8 * 256, 256, 0, stream>>>(y2, y1h, wd2, d2b);
  k_d3<<<CDIV(8 * 16384, 256), 256, 0, stream>>>(y3, y2, d3w, d3b);
  k_up<<<CDIV(8 * 512 * 512, 256), 256, 0, stream>>>(out, y3);
}
